// ConvEncoder_52802327937318
// MI455X (gfx1250) — compile-verified, matrix operand order checked by reading, not run on hardware
//
#include <hip/hip_runtime.h>
#include <hip/hip_bf16.h>

typedef _Float16 h16;
typedef __attribute__((ext_vector_type(16))) _Float16 v16h;
typedef __attribute__((ext_vector_type(8)))  _Float16 v8h;
typedef __attribute__((ext_vector_type(8)))  float    v8f;

union AFrag { v16h v; _Float16 e[16]; };
union CFrag { v8f  v; float     e[8];  };

#define BTC 2048   // B*T frames
#define TPB 256
#define USE_ASYNC_LDS 1

// ---------------------------------------------------------------------------
// A-fragment builder: KH (0 or 8) and all layer geometry compile-time, so
// k -> (ci,ky,kx) constant-folds; okm bit r9 = tap validity; yu/xu are the
// precomputed halved row/col indices for the nearest-2x upsampled source.
// ---------------------------------------------------------------------------
template<int KH, int Cin0, int Cin1, int lsh, int up0, int K>
__device__ __forceinline__ void fill_a(AFrag& a, int kb, unsigned okm,
                                       const h16* __restrict__ b0,
                                       const h16* __restrict__ b1,
                                       int y, int x,
                                       const int (&yu)[3], const int (&xu)[3])
{
    constexpr int W = 1 << lsh, HW = 1 << (2 * lsh), Wh = W >> 1, HWq = HW >> 2;
    #pragma unroll
    for (int e = 0; e < 16; ++e) {
        const int k = kb + KH + (e < 8 ? e : e + 8);
        _Float16 v = (_Float16)0.f;
        if (k < K) {
            const int ci = k / 9;
            const int r9 = k - ci * 9;       // tap id = ky*3+kx, ky/kx in 0..2
            const int ky = r9 / 3;
            const int kx = r9 - ky * 3;
            if ((okm >> r9) & 1u) {
                if (ci < Cin0) {
                    if (up0) v = b0[ci * HWq + yu[ky] * Wh + xu[kx]];
                    else     v = b0[ci * HW + (y + ky - 1) * W + (x + kx - 1)];
                } else {
                    v = b1[(ci - Cin0) * HW + (y + ky - 1) * W + (x + kx - 1)];
                }
            }
        }
        a.e[e] = v;
    }
}

// ---------------------------------------------------------------------------
// Implicit-GEMM 3x3 SAME conv via WMMA f16 (f32 acc). H == W == 1<<lsh.
// src0: [BT,Cin0,*]; if up0, read at (y>>1,x>>1). src1: skip concat channels.
// Weights are fragment-ordered and staged into LDS with async-to-LDS loads.
// One wave per 16-row M tile; loops (compile-time) N tiles.
// ---------------------------------------------------------------------------
template<int Cin0, int Cin1, int lsh, int up0, int KP, int NP, int Cout>
__global__ void __launch_bounds__(TPB)
conv_wmma_kernel(const h16* __restrict__ src0, const h16* __restrict__ src1,
                 const h16* __restrict__ wp, const float* __restrict__ bias,
                 h16* __restrict__ dst)
{
    constexpr int W   = 1 << lsh;
    constexpr int HW  = 1 << (2 * lsh);
    constexpr int HWq = HW >> 2;
    constexpr int Cin = Cin0 + Cin1;
    constexpr int K   = Cin * 9;
    constexpr int NT  = NP >> 4;
    constexpr int NCH = KP >> 5;

    __shared__ h16 wlds[KP * NP];

    // stage fragment-ordered weights into LDS (async DMA path, ASYNCcnt)
    for (int i = threadIdx.x * 8; i < KP * NP; i += TPB * 8) {
#if USE_ASYNC_LDS
        unsigned ldsoff = (unsigned)(size_t)(wlds + i);
        const h16* g = wp + i;
        asm volatile("global_load_async_to_lds_b128 %0, %1, off"
                     :: "v"(ldsoff), "v"(g) : "memory");
#else
        *(v8h*)(wlds + i) = *(const v8h*)(wp + i);
#endif
    }
#if USE_ASYNC_LDS
    asm volatile("s_wait_asynccnt 0x0" ::: "memory");
#endif
    __syncthreads();

    const int lane  = threadIdx.x & 31;
    const int wave  = threadIdx.x >> 5;
    const int mtile = blockIdx.x * (TPB >> 5) + wave;
    const int m0    = mtile << 4;

    // this lane's A-matrix row (fixed): decode frame/position once
    const int row = m0 + (lane & 15);
    const int f   = row >> (2 * lsh);
    const int rp  = row & (HW - 1);
    const int y   = rp >> lsh;
    const int x   = rp & (W - 1);
    const int khalf = (lane >> 4) << 3;          // 0 or 8

    // tap validity bitmask, bit r9 = (dy+1)*3 + (dx+1)
    unsigned okm = 0;
    #pragma unroll
    for (int dy = -1; dy <= 1; ++dy)
        #pragma unroll
        for (int dx = -1; dx <= 1; ++dx)
            if ((unsigned)(y + dy) < (unsigned)W && (unsigned)(x + dx) < (unsigned)W)
                okm |= 1u << ((dy + 1) * 3 + (dx + 1));

    const h16* b0 = src0 + (size_t)f * Cin0 * (up0 ? HWq : HW);
    const h16* b1 = src1 + (size_t)f * Cin1 * HW;
    int yu[3], xu[3];
    if (up0) {
        yu[0] = (y - 1) >> 1; yu[1] = y >> 1; yu[2] = (y + 1) >> 1;
        xu[0] = (x - 1) >> 1; xu[1] = x >> 1; xu[2] = (x + 1) >> 1;
    } else {
        yu[0] = yu[1] = yu[2] = 0; xu[0] = xu[1] = xu[2] = 0;
    }

    #pragma unroll
    for (int nt = 0; nt < NT; ++nt) {
        const int ncol = (nt << 4) + (lane & 15);
        CFrag acc;
        #pragma unroll
        for (int r = 0; r < 8; ++r) acc.e[r] = 0.f;

        #pragma unroll
        for (int c = 0; c < NCH; ++c) {
            AFrag a;
            if (lane < 16) fill_a<0, Cin0, Cin1, lsh, up0, K>(a, c << 5, okm, b0, b1, y, x, yu, xu);
            else           fill_a<8, Cin0, Cin1, lsh, up0, K>(a, c << 5, okm, b0, b1, y, x, yu, xu);
            const v16h bv = *(const v16h*)(wlds + (((c * NT + nt) << 9) + (lane << 4)));
            acc.v = __builtin_amdgcn_wmma_f32_16x16x32_f16(false, a.v, false, bv,
                                                           (short)0, acc.v, false, false);
        }

        if (ncol < Cout) {
            const float bn = bias[ncol];
            #pragma unroll
            for (int r = 0; r < 8; ++r) {
                const int mr = m0 + r + khalf;
                const int fr = mr >> (2 * lsh);
                const int rr = mr & (HW - 1);            // = y*W + x directly
                float v = fmaxf(acc.e[r] + bn, 0.f);
                dst[((size_t)(fr * Cout + ncol) << (2 * lsh)) + rr] = (h16)v;
            }
        }
    }
}

// ---------------------------------------------------------------------------
// Dense GEMM via WMMA: C = relu(A[M x lda] * B + bias).
// A rows contiguous f16 -> two 16B loads per 32-K chunk; B fragment-packed
// -> one v16h (2x b128) load per chunk. One wave per (mtile, ntile).
// ---------------------------------------------------------------------------
__global__ void gemm_wmma_kernel(const h16* __restrict__ A, int lda,
                                 const h16* __restrict__ Bp,
                                 const float* __restrict__ biasP,
                                 h16* __restrict__ C, int ldc,
                                 int KP, int NT, int relu)
{
    const int lane  = threadIdx.x & 31;
    const int wave  = threadIdx.x >> 5;
    const int wg    = blockIdx.x * (blockDim.x >> 5) + wave;
    const int mtile = wg / NT;
    const int ntile = wg - mtile * NT;
    const int m0    = mtile << 4;
    const int khalf = (lane >> 4) << 3;
    const int ncol  = (ntile << 4) + (lane & 15);

    CFrag acc;
    #pragma unroll
    for (int r = 0; r < 8; ++r) acc.e[r] = 0.f;

    const h16* arow = A + (size_t)(m0 + (lane & 15)) * lda;
    const h16* bptr = Bp + ((size_t)ntile << 9) + (lane << 4);
    for (int kb = 0; kb < KP; kb += 32) {
        v8h lo = *(const v8h*)(arow + kb + khalf);
        v8h hi = *(const v8h*)(arow + kb + khalf + 16);
        v16h a = __builtin_shufflevector(lo, hi, 0,1,2,3,4,5,6,7,8,9,10,11,12,13,14,15);
        const v16h bv = *(const v16h*)(bptr + ((size_t)(kb >> 5) * NT << 9));
        acc.v = __builtin_amdgcn_wmma_f32_16x16x32_f16(false, a, false, bv,
                                                       (short)0, acc.v, false, false);
    }
    const float bn = biasP[ncol];
    #pragma unroll
    for (int r = 0; r < 8; ++r) {
        float v = acc.e[r] + bn;
        if (relu) v = fmaxf(v, 0.f);
        C[(size_t)(m0 + r + khalf) * ldc + ncol] = (h16)v;
    }
}

// ---------------------------------------------------------------------------
// Glue kernels
// ---------------------------------------------------------------------------
__global__ void reorder_x_kernel(const float* __restrict__ x, h16* __restrict__ xt, int total)
{
    int i = blockIdx.x * blockDim.x + threadIdx.x;
    if (i >= total) return;
    int p  = i & 1023;
    int fc = i >> 10;
    int c  = fc % 3;
    int f  = fc / 3;
    int b  = f >> 6, t = f & 63;
    int h  = p >> 5, w = p & 31;
    size_t xi = ((((size_t)b * 3 + c) * 32 + h) * 32 + w) * 64 + t;   // [B,C,H,W,T]
    xt[i] = (h16)x[xi];
}

__global__ void pool_kernel(const h16* __restrict__ src, h16* __restrict__ dst,
                            int losh, int total)   // losh = log2(out W)
{
    int i = blockIdx.x * blockDim.x + threadIdx.x;
    if (i >= total) return;
    int xo = i & ((1 << losh) - 1);
    int t1 = i >> losh;
    int yo = t1 & ((1 << losh) - 1);
    int fc = t1 >> losh;                    // f*C + c
    int Wi = 2 << losh;
    const h16* s = src + (((size_t)fc << losh) * 2 + yo * 2) * Wi + xo * 2;
    float a = (float)s[0], b = (float)s[1], c = (float)s[Wi], d = (float)s[Wi + 1];
    dst[i] = (h16)fmaxf(fmaxf(a, b), fmaxf(c, d));
}

// 1x1 out-conv -> softmax over {l0,l1,1} -> masks * x -> feat [BT,6144] f16
__global__ void mask_feat_kernel(const h16* __restrict__ d1, const h16* __restrict__ xt,
                                 const float* __restrict__ ow, const float* __restrict__ ob,
                                 h16* __restrict__ feat, int total)
{
    int i = blockIdx.x * blockDim.x + threadIdx.x;
    if (i >= total) return;                 // total = BT*1024
    int p = i & 1023, f = i >> 10;
    float l0 = ob[0], l1 = ob[1];
    const h16* dp = d1 + (size_t)f * 8 * 1024 + p;
    #pragma unroll
    for (int c = 0; c < 8; ++c) {
        float dv = (float)dp[c * 1024];
        l0 += dv * ow[c];
        l1 += dv * ow[8 + c];
    }
    float mx = fmaxf(fmaxf(l0, l1), 1.0f);
    float e0 = expf(l0 - mx), e1 = expf(l1 - mx), e2 = expf(1.0f - mx);
    float inv = 1.0f / (e0 + e1 + e2);
    float m0 = e0 * inv, m1 = e1 * inv;
    size_t fb = (size_t)f * 6144;
    #pragma unroll
    for (int c = 0; c < 3; ++c) {
        float xv = (float)xt[((size_t)f * 3 + c) * 1024 + p];
        feat[fb + c * 1024 + p]        = (h16)(m0 * xv);
        feat[fb + 3072 + c * 1024 + p] = (h16)(m1 * xv);
    }
}

__global__ void mlp3_kernel(const h16* __restrict__ h2, const float* __restrict__ w3,
                            const float* __restrict__ b3, float* __restrict__ out, int total)
{
    int i = blockIdx.x * blockDim.x + threadIdx.x;
    if (i >= total) return;                 // total = BT*4
    int j = i & 3, f = i >> 2;
    const h16* hp = h2 + (size_t)f * 224;
    float s = b3[j];
    for (int k = 0; k < 200; ++k) s += (float)hp[k] * w3[k * 4 + j];
    out[i] = tanhf(s) * 16.0f + 16.0f;
}

// ---------------------------------------------------------------------------
// Fragment-order weight packers.
// dst index i: e = i&15, lane = (i>>4)&31, nt = (i>>9)%NT, cb = (i>>9)/NT
//   k = cb*32 + ((lane>>4)<<3) + (e<8 ? e : e+8);  n = nt*16 + (lane&15)
// ---------------------------------------------------------------------------
__global__ void pack_conv_frag(const float* __restrict__ w, int Cout, int Cin,
                               int NT, int total, h16* __restrict__ dst)
{
    int i = blockIdx.x * blockDim.x + threadIdx.x;
    if (i >= total) return;
    int e    = i & 15;
    int lane = (i >> 4) & 31;
    int tn   = i >> 9;
    int nt   = tn % NT;
    int cb   = tn / NT;
    int k    = cb * 32 + ((lane >> 4) << 3) + (e < 8 ? e : e + 8);
    int n    = (nt << 4) + (lane & 15);
    float v  = 0.f;
    if (n < Cout && k < Cin * 9) {
        int ci = k / 9, r = k - ci * 9;
        v = w[((size_t)n * Cin + ci) * 9 + r];
    }
    dst[i] = (h16)v;
}

__global__ void pack_fc_frag(const float* __restrict__ w, int Kr, int Nr,
                             int NT, int total, h16* __restrict__ dst)
{
    int i = blockIdx.x * blockDim.x + threadIdx.x;
    if (i >= total) return;
    int e    = i & 15;
    int lane = (i >> 4) & 31;
    int tn   = i >> 9;
    int nt   = tn % NT;
    int cb   = tn / NT;
    int k    = cb * 32 + ((lane >> 4) << 3) + (e < 8 ? e : e + 8);
    int n    = (nt << 4) + (lane & 15);
    dst[i] = (h16)((k < Kr && n < Nr) ? w[(size_t)k * Nr + n] : 0.f);
}

__global__ void pack_bias_kernel(const float* __restrict__ b1, const float* __restrict__ b2,
                                 float* __restrict__ b1p, float* __restrict__ b2p)
{
    int i = blockIdx.x * blockDim.x + threadIdx.x;
    if (i >= 208) return;
    b1p[i] = (i < 200) ? b1[i] : 0.f;
    b2p[i] = (i < 200) ? b2[i] : 0.f;
}

__global__ void zero_h_kernel(h16* __restrict__ h1, h16* __restrict__ h2, int total)
{
    int i = blockIdx.x * blockDim.x + threadIdx.x;
    if (i >= total) return;
    h1[i] = (h16)0.f;
    h2[i] = (h16)0.f;
}

// ---------------------------------------------------------------------------
extern "C" void kernel_launch(void* const* d_in, const int* in_sizes, int n_in,
                              void* d_out, int out_size, void* d_ws, size_t ws_size,
                              hipStream_t stream)
{
    const float* x      = (const float*)d_in[0];
    const float* enc_w1 = (const float*)d_in[1];
    const float* enc_b1 = (const float*)d_in[2];
    const float* enc_w2 = (const float*)d_in[3];
    const float* enc_b2 = (const float*)d_in[4];
    const float* enc_w3 = (const float*)d_in[5];
    const float* enc_b3 = (const float*)d_in[6];
    const float* dec_w2 = (const float*)d_in[7];
    const float* dec_b2 = (const float*)d_in[8];
    const float* dec_w1 = (const float*)d_in[9];
    const float* dec_b1 = (const float*)d_in[10];
    const float* out_w  = (const float*)d_in[11];
    const float* out_b  = (const float*)d_in[12];
    const float* loc_w1 = (const float*)d_in[13];
    const float* loc_b1 = (const float*)d_in[14];
    const float* loc_w2 = (const float*)d_in[15];
    const float* loc_b2 = (const float*)d_in[16];
    const float* loc_w3 = (const float*)d_in[17];
    const float* loc_b3 = (const float*)d_in[18];
    float* out = (float*)d_out;

    char* base = (char*)d_ws;
    size_t off = 0;
    auto alloc = [&](size_t bytes) -> char* {
        char* p = base + off;
        off += (bytes + 255) & ~(size_t)255;
        return p;
    };

    h16* xt   = (h16*)alloc((size_t)BTC * 3  * 1024 * 2);
    h16* e1   = (h16*)alloc((size_t)BTC * 8  * 1024 * 2);
    h16* p1   = (h16*)alloc((size_t)BTC * 8  * 256  * 2);
    h16* e2   = (h16*)alloc((size_t)BTC * 16 * 256  * 2);
    h16* p2   = (h16*)alloc((size_t)BTC * 16 * 64   * 2);
    h16* e3   = (h16*)alloc((size_t)BTC * 32 * 64   * 2);
    h16* d2   = (h16*)alloc((size_t)BTC * 16 * 256  * 2);
    h16* d1   = (h16*)alloc((size_t)BTC * 8  * 1024 * 2);
    h16* feat = (h16*)alloc((size_t)BTC * 6144 * 2);
    h16* h1   = (h16*)alloc((size_t)BTC * 224 * 2);
    h16* h2   = (h16*)alloc((size_t)BTC * 224 * 2);
    h16* wpe1 = (h16*)alloc(32  * 16 * 2);
    h16* wpe2 = (h16*)alloc(96  * 16 * 2);
    h16* wpe3 = (h16*)alloc(160 * 32 * 2);
    h16* wpd2 = (h16*)alloc(448 * 16 * 2);
    h16* wpd1 = (h16*)alloc(224 * 16 * 2);
    h16* w1p  = (h16*)alloc((size_t)6144 * 208 * 2);
    h16* w2p  = (h16*)alloc((size_t)224  * 208 * 2);
    float* b1p = (float*)alloc(208 * 4);
    float* b2p = (float*)alloc(208 * 4);

    auto cdiv = [](long n, int b) { return (int)((n + b - 1) / b); };

    // --- prep: pack weights (fragment order) / biases, zero padded h buffers ---
    pack_conv_frag<<<cdiv(32 * 16, TPB),  TPB, 0, stream>>>(enc_w1,  8,  3, 1, 32 * 16,  wpe1);
    pack_conv_frag<<<cdiv(96 * 16, TPB),  TPB, 0, stream>>>(enc_w2, 16,  8, 1, 96 * 16,  wpe2);
    pack_conv_frag<<<cdiv(160 * 32, TPB), TPB, 0, stream>>>(enc_w3, 32, 16, 2, 160 * 32, wpe3);
    pack_conv_frag<<<cdiv(448 * 16, TPB), TPB, 0, stream>>>(dec_w2, 16, 48, 1, 448 * 16, wpd2);
    pack_conv_frag<<<cdiv(224 * 16, TPB), TPB, 0, stream>>>(dec_w1,  8, 24, 1, 224 * 16, wpd1);
    pack_fc_frag<<<cdiv((long)6144 * 208, TPB), TPB, 0, stream>>>(loc_w1, 6144, 200, 13,
                                                                  6144 * 208, w1p);
    pack_fc_frag<<<cdiv((long)224 * 208,  TPB), TPB, 0, stream>>>(loc_w2,  200, 200, 13,
                                                                  224 * 208,  w2p);
    pack_bias_kernel<<<1, TPB, 0, stream>>>(loc_b1, loc_b2, b1p, b2p);
    zero_h_kernel<<<cdiv((long)BTC * 224, TPB), TPB, 0, stream>>>(h1, h2, BTC * 224);

    // --- input reorder [B,C,H,W,T] -> [BT,C,H,W] f16 ---
    reorder_x_kernel<<<cdiv((long)BTC * 3 * 1024, TPB), TPB, 0, stream>>>(x, xt, BTC * 3 * 1024);

    // --- UNet: all convs as templated WMMA implicit GEMM; 8 waves/block ---
    // enc1: 3->8 @32x32, Mtiles = 131072
    conv_wmma_kernel<3, 0, 5, 0, 32, 16, 8>
        <<<131072 / 8, TPB, 0, stream>>>(xt, nullptr, wpe1, enc_b1, e1);
    pool_kernel<<<cdiv((long)BTC * 8 * 256, TPB), TPB, 0, stream>>>(e1, p1, 4, BTC * 8 * 256);
    // enc2: 8->16 @16x16, Mtiles = 32768
    conv_wmma_kernel<8, 0, 4, 0, 96, 16, 16>
        <<<32768 / 8, TPB, 0, stream>>>(p1, nullptr, wpe2, enc_b2, e2);
    pool_kernel<<<cdiv((long)BTC * 16 * 64, TPB), TPB, 0, stream>>>(e2, p2, 3, BTC * 16 * 64);
    // enc3: 16->32 @8x8, Mtiles = 8192
    conv_wmma_kernel<16, 0, 3, 0, 160, 32, 32>
        <<<8192 / 8, TPB, 0, stream>>>(p2, nullptr, wpe3, enc_b3, e3);
    // dec2: concat(up(e3)[32ch], e2[16ch]) -> 16 @16x16, Mtiles = 32768
    conv_wmma_kernel<32, 16, 4, 1, 448, 16, 16>
        <<<32768 / 8, TPB, 0, stream>>>(e3, e2, wpd2, dec_b2, d2);
    // dec1: concat(up(d2)[16ch], e1[8ch]) -> 8 @32x32, Mtiles = 131072
    conv_wmma_kernel<16, 8, 5, 1, 224, 16, 8>
        <<<131072 / 8, TPB, 0, stream>>>(d2, e1, wpd1, dec_b1, d1);

    // --- 1x1 conv + softmax + mask*x -> feat ---
    mask_feat_kernel<<<cdiv((long)BTC * 1024, TPB), TPB, 0, stream>>>(d1, xt, out_w, out_b,
                                                                      feat, BTC * 1024);

    // --- LocationNetwork GEMMs: 128 Mtiles x 13 Ntiles = 1664 waves each ---
    gemm_wmma_kernel<<<1664 / 8, TPB, 0, stream>>>(feat, 6144, w1p, b1p, h1, 224, 6144, 13, 1);
    gemm_wmma_kernel<<<1664 / 8, TPB, 0, stream>>>(h1,   224,  w2p, b2p, h2, 224,  224, 13, 1);

    // --- head: 200->4, tanh*16+16 -> out [B,T,2,2] ---
    mlp3_kernel<<<cdiv((long)BTC * 4, TPB), TPB, 0, stream>>>(h2, loc_w3, loc_b3, out, BTC * 4);
}
